// MultiRelationGE_GNNLayer_85512798863506
// MI455X (gfx1250) — matrix operand order repaired and checked by reading, not compiled, not run on hardware
//
#include <hip/hip_runtime.h>
#include <math.h>

// Problem constants (match reference)
#define N_NODES 100000
#define E_EDGES 1200000
#define R_REL   3
#define D_INF   64
#define HEAD    4
#define HD      16
#define HHD     64   // HEAD*HD
#define NEG_SLOPE 0.01f

typedef __attribute__((ext_vector_type(16))) _Float16 v16h;
typedef __attribute__((ext_vector_type(8)))  float    v8f;

// ---- WMMA fragment index helpers (CDNA5 ISA 7.12.2, wave32) ----
// 16-bit A-matrix 16x32: lanes 0-15 -> M=0..15, K in {0..7,16..23};
// lanes 16-31 -> same M, K in {8..15,24..31}. Element e of v16h maps to
// VGPR e>>1, halfword e&1.
__device__ __forceinline__ int wmma_a_k(int e, int half) {
  int v = e >> 1, p = e & 1;
  int base = (v < 4) ? (2 * v) : (16 + 2 * (v - 4));
  return base + p + 8 * half;
}
// B-matrix 32x16: lane holds column c=lane&15; lanes 0-15 K in {0..7,16..23},
// lanes 16-31 K in {8..15,24..31} (same K order as A).
__device__ __forceinline__ int wmma_b_k(int e, int half) {
  return (e < 8) ? (e + 8 * half) : (16 + 8 * half + (e - 8));
}

// Monotone float<->uint encoding for atomicMax-based segment max
__device__ __forceinline__ unsigned encf(float xv) {
  unsigned u = __float_as_uint(xv);
  return (u & 0x80000000u) ? ~u : (u | 0x80000000u);
}
__device__ __forceinline__ float decf(unsigned u) {
  return (u & 0x80000000u) ? __uint_as_float(u & 0x7fffffffu)
                           : __uint_as_float(~u);
}

// =====================================================================
// K1: node transforms via WMMA with LDS-staged, fragment-packed weights.
//   esn = tanh(2*(h @ d_w + d_b))          [N, 64]
//   hp[r] = h @ w_w[r] + w_b[r], r=0..2    [3, N, 64]
// 8 waves/block, one 16-row tile per wave; weights packed once per block
// into LDS in exact B-fragment order -> each fragment = one 32B ds read.
// =====================================================================
__global__ void __launch_bounds__(256)
node_transform_wmma(const float* __restrict__ h,
                    const float* __restrict__ d_w,
                    const float* __restrict__ d_b,
                    const float* __restrict__ w_w,
                    const float* __restrict__ w_b,
                    float* __restrict__ esn,
                    float* __restrict__ hp) {
  // packed: [w(4)][kc(2)][ct(4)][lane(32)][e(16)] halves = 32 KB
  __shared__ __align__(32) _Float16 wpack[4 * 2 * 4 * 32 * 16];

  // ---- cooperative weight staging (all threads, before any exit) ----
  for (int idx = threadIdx.x; idx < 4 * 2 * 4 * 32 * 16; idx += blockDim.x) {
    const int e    = idx & 15;
    const int lane = (idx >> 4) & 31;
    const int ct   = (idx >> 9) & 3;
    const int kc   = (idx >> 11) & 1;
    const int w    = (idx >> 12) & 3;
    const int half = lane >> 4;
    const int c    = ct * 16 + (lane & 15);
    const int k    = 32 * kc + wmma_b_k(e, half);
    const float* W = (w == 0) ? d_w : (w_w + (size_t)(w - 1) * D_INF * HHD);
    wpack[idx] = (_Float16)W[(size_t)k * HHD + c];
  }
  __syncthreads();

  const int lane = threadIdx.x & 31;
  const int wave = threadIdx.x >> 5;
  const int tile = blockIdx.x * (blockDim.x >> 5) + wave;
  const int numTiles = (N_NODES + 15) / 16;
  if (tile >= numTiles) return;           // wave-uniform exit (EXEC stays all-1)
  const int row0 = tile * 16;
  const int half = lane >> 4;
  const int rl   = lane & 15;
  int row = row0 + rl; if (row >= N_NODES) row = N_NODES - 1;

  // A fragments for K-chunks [0..31], [32..63] (f32 -> f16)
  v16h a[2];
#pragma unroll
  for (int kc = 0; kc < 2; ++kc) {
    const float* p = h + (size_t)row * D_INF + 32 * kc;
#pragma unroll
    for (int e = 0; e < 16; ++e) a[kc][e] = (_Float16)p[wmma_a_k(e, half)];
  }

#pragma unroll
  for (int w = 0; w < 4; ++w) {
    const float* B = (w == 0) ? d_b : (w_b + (size_t)(w - 1) * HHD);
#pragma unroll
    for (int ct = 0; ct < 4; ++ct) {
      const int c = ct * 16 + rl;
      const float bias = B[c];
      v8f acc;
#pragma unroll
      for (int i = 0; i < 8; ++i) acc[i] = bias;
#pragma unroll
      for (int kc = 0; kc < 2; ++kc) {
        const v16h bf = *(const v16h*)&wpack[(((w * 2 + kc) * 4 + ct) * 32 + lane) * 16];
        acc = __builtin_amdgcn_wmma_f32_16x16x32_f16(
            false, a[kc], false, bf, (short)0, acc, false, false);
      }
#pragma unroll
      for (int i = 0; i < 8; ++i) {
        const int m = row0 + i + 8 * half;
        if (m < N_NODES) {
          if (w == 0)
            esn[(size_t)m * HHD + c] = tanhf(2.0f * acc[i]);
          else
            hp[(size_t)(w - 1) * N_NODES * HHD + (size_t)m * HHD + c] = acc[i];
        }
      }
    }
  }
}

// =====================================================================
// Per-relation accumulator init
// =====================================================================
__global__ void init_accums(unsigned* __restrict__ amax,
                            float* __restrict__ denom,
                            float* __restrict__ S,
                            float* __restrict__ es_sum,
                            float* __restrict__ cnt) {
  const int i = blockIdx.x * blockDim.x + threadIdx.x;
  if (i < N_NODES * HHD) { S[i] = 0.0f; es_sum[i] = 0.0f; }
  if (i < N_NODES * HEAD) { amax[i] = 0x007FFFFFu /* enc(-inf) */; denom[i] = 0.0f; }
  if (i < N_NODES) cnt[i] = 0.0f;
}

// =====================================================================
// K2: per-edge attention logits + leaky ReLU + per-(dst,head) max
// =====================================================================
__global__ void edge_logits(const int* __restrict__ src,
                            const int* __restrict__ dst,
                            const float* __restrict__ hp_r,
                            const float* __restrict__ esn,
                            const float* __restrict__ atten_w,  // all R, [R,48]
                            const float* __restrict__ atten_b,  // all R, [R]
                            int r,
                            float* __restrict__ alpha,
                            unsigned* __restrict__ amax) {
  const int e = blockIdx.x * blockDim.x + threadIdx.x;
  if (e >= E_EDGES) return;
  const int s = src[e], t = dst[e];
  const float4* sh = (const float4*)(hp_r + (size_t)s * HHD);
  const float4* th = (const float4*)(hp_r + (size_t)t * HHD);
  const float4* es = (const float4*)(esn  + (size_t)s * HHD);
  const float* wa = atten_w + (size_t)r * 3 * HD;
  const float  ab = atten_b[r];
#pragma unroll
  for (int hh = 0; hh < HEAD; ++hh) {
    float lg = ab;
#pragma unroll
    for (int q = 0; q < HD / 4; ++q) {
      const float4 sv = sh[hh * 4 + q];
      const float4 tv = th[hh * 4 + q];
      const float4 ev = es[hh * 4 + q];
      const int d = q * 4;
      lg += sv.x * wa[d]     + tv.x * wa[HD + d]     + ev.x * wa[2 * HD + d];
      lg += sv.y * wa[d + 1] + tv.y * wa[HD + d + 1] + ev.y * wa[2 * HD + d + 1];
      lg += sv.z * wa[d + 2] + tv.z * wa[HD + d + 2] + ev.z * wa[2 * HD + d + 2];
      lg += sv.w * wa[d + 3] + tv.w * wa[HD + d + 3] + ev.w * wa[2 * HD + d + 3];
    }
    const float al = (lg >= 0.0f) ? lg : NEG_SLOPE * lg;
    alpha[(size_t)e * HEAD + hh] = al;
    atomicMax(&amax[t * HEAD + hh], encf(al));
  }
}

// =====================================================================
// K3: ex = exp(alpha - max); accumulate denom, S = sum(ex*src_h),
//     es_sum = sum(esn[s]), cnt
// =====================================================================
__global__ void edge_aggregate(const int* __restrict__ src,
                               const int* __restrict__ dst,
                               const float* __restrict__ hp_r,
                               const float* __restrict__ esn,
                               const float* __restrict__ alpha,
                               const unsigned* __restrict__ amax,
                               float* __restrict__ denom,
                               float* __restrict__ S,
                               float* __restrict__ es_sum,
                               float* __restrict__ cnt) {
  const int e = blockIdx.x * blockDim.x + threadIdx.x;
  if (e >= E_EDGES) return;
  const int s = src[e], t = dst[e];
  float ex[HEAD];
#pragma unroll
  for (int hh = 0; hh < HEAD; ++hh) {
    const float m = decf(amax[t * HEAD + hh]);
    ex[hh] = expf(alpha[(size_t)e * HEAD + hh] - m);
    atomicAdd(&denom[t * HEAD + hh], ex[hh]);
  }
  const float4* sh = (const float4*)(hp_r + (size_t)s * HHD);
  const float4* es = (const float4*)(esn  + (size_t)s * HHD);
#pragma unroll
  for (int q = 0; q < HHD / 4; ++q) {
    const float4 sv = sh[q];
    const float4 ev = es[q];
    const float  xh = ex[q >> 2];     // q*4 >> 4
    float* Sp = S      + (size_t)t * HHD + q * 4;
    float* Ep = es_sum + (size_t)t * HHD + q * 4;
    atomicAdd(Sp + 0, xh * sv.x);  atomicAdd(Ep + 0, ev.x);
    atomicAdd(Sp + 1, xh * sv.y);  atomicAdd(Ep + 1, ev.y);
    atomicAdd(Sp + 2, xh * sv.z);  atomicAdd(Ep + 2, ev.z);
    atomicAdd(Sp + 3, xh * sv.w);  atomicAdd(Ep + 3, ev.w);
  }
  atomicAdd(&cnt[t], 1.0f);
}

// =====================================================================
// K4: per-node normalize + gate; write x[:, r*64 : (r+1)*64]
// =====================================================================
__global__ void node_gate(const float* __restrict__ S,
                          const float* __restrict__ denom,
                          const float* __restrict__ es_sum,
                          const float* __restrict__ cnt,
                          const float* __restrict__ hp_r,
                          const float* __restrict__ beta, // all R, [R,128]
                          int r,
                          float* __restrict__ x /* [N, R*HHD] */) {
  const int n = blockIdx.x * blockDim.x + threadIdx.x;
  if (n >= N_NODES) return;
  const float* br = beta + (size_t)r * 2 * HHD;
  const float c = fmaxf(cnt[n], 1.0f);
  float out[HHD];
  float g = 0.0f;
#pragma unroll 8
  for (int d = 0; d < HHD; ++d) {
    const float den = fmaxf(denom[n * HEAD + (d >> 4)], 1e-20f);
    const float o = S[(size_t)n * HHD + d] / den;
    out[d] = o;
    const float en = es_sum[(size_t)n * HHD + d] / c;
    g += br[d] * en + br[HHD + d] * o;
  }
  const float gate = 1.0f / (1.0f + expf(-g));
#pragma unroll 8
  for (int d = 0; d < HHD; ++d) {
    x[(size_t)n * (R_REL * HHD) + (size_t)r * HHD + d] =
        gate * out[d] + (1.0f - gate) * hp_r[(size_t)n * HHD + d];
  }
}

// =====================================================================
// K5: final linear via WMMA: out = x[N,192] @ lin_w[192,64] + lin_b
// LDS-staged fragment-packed lin_w (24 KB).
// =====================================================================
__global__ void __launch_bounds__(256)
final_linear_wmma(const float* __restrict__ x,
                  const float* __restrict__ lin_w,
                  const float* __restrict__ lin_b,
                  float* __restrict__ out) {
  // packed: [kc(6)][ct(4)][lane(32)][e(16)] halves = 24 KB
  __shared__ __align__(32) _Float16 wpack[6 * 4 * 32 * 16];

  for (int idx = threadIdx.x; idx < 6 * 4 * 32 * 16; idx += blockDim.x) {
    const int e    = idx & 15;
    const int lane = (idx >> 4) & 31;
    const int ct   = (idx >> 9) & 3;
    const int kc   = (idx >> 11);        // 0..5
    const int half = lane >> 4;
    const int c    = ct * 16 + (lane & 15);
    const int k    = 32 * kc + wmma_b_k(e, half);
    wpack[idx] = (_Float16)lin_w[(size_t)k * HHD + c];
  }
  __syncthreads();

  const int lane = threadIdx.x & 31;
  const int wave = threadIdx.x >> 5;
  const int tile = blockIdx.x * (blockDim.x >> 5) + wave;
  const int numTiles = (N_NODES + 15) / 16;
  if (tile >= numTiles) return;
  const int row0 = tile * 16;
  const int half = lane >> 4;
  const int rl   = lane & 15;
  int row = row0 + rl; if (row >= N_NODES) row = N_NODES - 1;
  const int KDIM = R_REL * HHD;  // 192

  v16h a[6];
#pragma unroll
  for (int kc = 0; kc < 6; ++kc) {
    const float* p = x + (size_t)row * KDIM + 32 * kc;
#pragma unroll
    for (int e = 0; e < 16; ++e) a[kc][e] = (_Float16)p[wmma_a_k(e, half)];
  }

#pragma unroll
  for (int ct = 0; ct < 4; ++ct) {
    const int c = ct * 16 + rl;
    const float bias = lin_b[c];
    v8f acc;
#pragma unroll
    for (int i = 0; i < 8; ++i) acc[i] = bias;
#pragma unroll
    for (int kc = 0; kc < 6; ++kc) {
      const v16h bf = *(const v16h*)&wpack[((kc * 4 + ct) * 32 + lane) * 16];
      acc = __builtin_amdgcn_wmma_f32_16x16x32_f16(
          false, a[kc], false, bf, (short)0, acc, false, false);
    }
#pragma unroll
    for (int i = 0; i < 8; ++i) {
      const int m = row0 + i + 8 * half;
      if (m < N_NODES) out[(size_t)m * HHD + c] = acc[i];
    }
  }
}

// =====================================================================
extern "C" void kernel_launch(void* const* d_in, const int* in_sizes, int n_in,
                              void* d_out, int out_size, void* d_ws, size_t ws_size,
                              hipStream_t stream) {
  const float* h       = (const float*)d_in[0];
  const int*   src     = (const int*)  d_in[1];
  const int*   dst     = (const int*)  d_in[2];
  const float* d_w     = (const float*)d_in[3];
  const float* d_b     = (const float*)d_in[4];
  const float* w_w     = (const float*)d_in[5];
  const float* w_b     = (const float*)d_in[6];
  const float* atten_w = (const float*)d_in[7];
  const float* atten_b = (const float*)d_in[8];
  const float* beta    = (const float*)d_in[9];
  const float* lin_w   = (const float*)d_in[10];
  const float* lin_b   = (const float*)d_in[11];
  float* out = (float*)d_out;

  char* ws = (char*)d_ws;
  size_t off = 0;
  auto alloc = [&](size_t bytes) -> void* {
    void* p = ws + off;
    off += (bytes + 255) & ~(size_t)255;
    return p;
  };
  float*    esn    = (float*)   alloc((size_t)N_NODES * HHD * 4);
  float*    hp     = (float*)   alloc((size_t)R_REL * N_NODES * HHD * 4);
  float*    x      = (float*)   alloc((size_t)N_NODES * R_REL * HHD * 4);
  float*    alpha  = (float*)   alloc((size_t)E_EDGES * HEAD * 4);
  unsigned* amax   = (unsigned*)alloc((size_t)N_NODES * HEAD * 4);
  float*    denom  = (float*)   alloc((size_t)N_NODES * HEAD * 4);
  float*    S      = (float*)   alloc((size_t)N_NODES * HHD * 4);
  float*    es_sum = (float*)   alloc((size_t)N_NODES * HHD * 4);
  float*    cnt    = (float*)   alloc((size_t)N_NODES * 4);
  (void)ws_size; (void)in_sizes; (void)n_in; (void)out_size;

  const int tiles = (N_NODES + 15) / 16;
  const int gemmBlocks = (tiles + 7) / 8;          // 8 waves (256 thr) per block

  node_transform_wmma<<<gemmBlocks, 256, 0, stream>>>(h, d_w, d_b, w_w, w_b,
                                                      esn, hp);
  for (int r = 0; r < R_REL; ++r) {
    const float* hp_r  = hp + (size_t)r * N_NODES * HHD;
    const int*   src_r = src + (size_t)r * E_EDGES;
    const int*   dst_r = dst + (size_t)r * E_EDGES;
    init_accums<<<(N_NODES * HHD + 255) / 256, 256, 0, stream>>>(
        amax, denom, S, es_sum, cnt);
    edge_logits<<<(E_EDGES + 255) / 256, 256, 0, stream>>>(
        src_r, dst_r, hp_r, esn, atten_w, atten_b, r, alpha, amax);
    edge_aggregate<<<(E_EDGES + 255) / 256, 256, 0, stream>>>(
        src_r, dst_r, hp_r, esn, alpha, amax, denom, S, es_sum, cnt);
    node_gate<<<(N_NODES + 255) / 256, 256, 0, stream>>>(
        S, denom, es_sum, cnt, hp_r, beta, r, x);
  }
  final_linear_wmma<<<gemmBlocks, 256, 0, stream>>>(x, lin_w, lin_b, out);
}
